// Seq2Seq_75153337745711
// MI455X (gfx1250) — matrix operand order repaired and checked by reading, not compiled, not run on hardware
//
#include <hip/hip_runtime.h>
#include <math.h>

// Problem constants (match reference)
#define BB   32
#define SS   64
#define TT   64
#define WVD  512
#define HH   1024
#define HD2  512        // per-direction encoder hidden
#define VV   32000

typedef __attribute__((ext_vector_type(16))) _Float16 v16h;
typedef __attribute__((ext_vector_type(8)))  float    v8f;

__device__ __forceinline__ void cvt8(v16h& dst, int base, float4 x, float4 y)
{
    dst[base + 0] = (_Float16)x.x;
    dst[base + 1] = (_Float16)x.y;
    dst[base + 2] = (_Float16)x.z;
    dst[base + 3] = (_Float16)x.w;
    dst[base + 4] = (_Float16)y.x;
    dst[base + 5] = (_Float16)y.y;
    dst[base + 6] = (_Float16)y.z;
    dst[base + 7] = (_Float16)y.w;
}

// ---------------------------------------------------------------------------
// WMMA GEMM:  C[M,N] = A[M,K] @ W[N,K]^T  (+ bias[N]) (+ C if accum)
// A row-major f32 (lda), W row-major f32 in PyTorch Linear layout (ldw).
// One wave computes a 16x64 strip (1x4 register-blocked 16x16 tiles): the A
// fragment is loaded once and reused by 4 v_wmma_f32_16x16x32_f16.
// ALL 20 float4 loads of a K-iteration are hoisted into distinct live values
// before any conversion/WMMA so the backend issues them as one load clause
// (high memory-level parallelism) instead of load->wait->load chains.
// Requires: M % 16 == 0, N % 64 == 0, K % 32 == 0, 16B-aligned rows.
// ---------------------------------------------------------------------------
__global__ __launch_bounds__(256) void k_gemm(
    const float* __restrict__ A, int lda,
    const float* __restrict__ W, int ldw,
    const float* __restrict__ bias,
    float* __restrict__ C, long ldc,
    int N, int K, int accum, int totalTiles)
{
    const int lane = threadIdx.x & 31;
    const int wave = threadIdx.x >> 5;
    const int tile = blockIdx.x * 8 + wave;       // 16x64 strip index
    if (tile >= totalTiles) return;
    const int strips = N >> 6;
    const int tm = tile / strips;
    const int tn4 = tile - tm * strips;
    const int hi = lane >> 4;     // lane half (K-group selector)
    const int lo = lane & 15;     // A row / B column within tile

    const float* arow  = A + (size_t)(tm * 16 + lo) * (size_t)lda;
    const float* wrow0 = W + (size_t)(tn4 * 64 + 0 * 16 + lo) * (size_t)ldw;
    const float* wrow1 = W + (size_t)(tn4 * 64 + 1 * 16 + lo) * (size_t)ldw;
    const float* wrow2 = W + (size_t)(tn4 * 64 + 2 * 16 + lo) * (size_t)ldw;
    const float* wrow3 = W + (size_t)(tn4 * 64 + 3 * 16 + lo) * (size_t)ldw;

    v8f acc0 = {}, acc1 = {}, acc2 = {}, acc3 = {};
    for (int k0 = 0; k0 < K; k0 += 32) {
        const float* ab = arow  + k0 + hi * 16;
        const float* w0 = wrow0 + k0 + hi * 16;
        const float* w1 = wrow1 + k0 + hi * 16;
        const float* w2 = wrow2 + k0 + hi * 16;
        const float* w3 = wrow3 + k0 + hi * 16;

        // ---- issue ALL loads first (distinct destinations, one clause) ----
        float4 a0 = *(const float4*)(ab + 0);
        float4 a1 = *(const float4*)(ab + 4);
        float4 a2 = *(const float4*)(ab + 8);
        float4 a3 = *(const float4*)(ab + 12);
        float4 b00 = *(const float4*)(w0 + 0);
        float4 b01 = *(const float4*)(w0 + 4);
        float4 b02 = *(const float4*)(w0 + 8);
        float4 b03 = *(const float4*)(w0 + 12);
        float4 b10 = *(const float4*)(w1 + 0);
        float4 b11 = *(const float4*)(w1 + 4);
        float4 b12 = *(const float4*)(w1 + 8);
        float4 b13 = *(const float4*)(w1 + 12);
        float4 b20 = *(const float4*)(w2 + 0);
        float4 b21 = *(const float4*)(w2 + 4);
        float4 b22 = *(const float4*)(w2 + 8);
        float4 b23 = *(const float4*)(w2 + 12);
        float4 b30 = *(const float4*)(w3 + 0);
        float4 b31 = *(const float4*)(w3 + 4);
        float4 b32 = *(const float4*)(w3 + 8);
        float4 b33 = *(const float4*)(w3 + 12);

        if (k0 + 32 < K) {
            __builtin_prefetch(arow + k0 + 32, 0, 0);   // global_prefetch_b8
            __builtin_prefetch(wrow0 + k0 + 32, 0, 0);
            __builtin_prefetch(wrow1 + k0 + 32, 0, 0);
            __builtin_prefetch(wrow2 + k0 + 32, 0, 0);
            __builtin_prefetch(wrow3 + k0 + 32, 0, 0);
        }

        // ---- pack to f16 fragments ----
        // A (16x32 f16 layout): k = k0 + hi*16 + e, row = lane&15
        v16h af;  cvt8(af, 0, a0, a1);   cvt8(af, 8, a2, a3);
        // B (32x16 f16 layout): k = k0 + hi*16 + e, col = lane&15
        v16h bf0; cvt8(bf0, 0, b00, b01); cvt8(bf0, 8, b02, b03);
        v16h bf1; cvt8(bf1, 0, b10, b11); cvt8(bf1, 8, b12, b13);
        v16h bf2; cvt8(bf2, 0, b20, b21); cvt8(bf2, 8, b22, b23);
        v16h bf3; cvt8(bf3, 0, b30, b31); cvt8(bf3, 8, b32, b33);

        acc0 = __builtin_amdgcn_wmma_f32_16x16x32_f16(false, af, false, bf0,
                                                      (short)0, acc0, false, false);
        acc1 = __builtin_amdgcn_wmma_f32_16x16x32_f16(false, af, false, bf1,
                                                      (short)0, acc1, false, false);
        acc2 = __builtin_amdgcn_wmma_f32_16x16x32_f16(false, af, false, bf2,
                                                      (short)0, acc2, false, false);
        acc3 = __builtin_amdgcn_wmma_f32_16x16x32_f16(false, af, false, bf3,
                                                      (short)0, acc3, false, false);
    }
    // Store: D layout row m = r + 8*half, col n = lane&15 within each 16x16 tile
    v8f accs[4] = {acc0, acc1, acc2, acc3};
#pragma unroll
    for (int s = 0; s < 4; ++s) {
        const int col = tn4 * 64 + s * 16 + lo;
        const float bv = bias ? bias[col] : 0.0f;
#pragma unroll
        for (int r = 0; r < 8; ++r) {
            const int row = tm * 16 + r + hi * 8;
            const size_t idx = (size_t)row * (size_t)ldc + (size_t)col;
            float v = accs[s][r] + bv;
            if (accum) v += C[idx];
            C[idx] = v;
        }
    }
}

// ---------------------------------------------------------------------------
// Elementwise / small kernels
// ---------------------------------------------------------------------------
__global__ void k_zero(float* __restrict__ p, int n)
{
    int i = blockIdx.x * blockDim.x + threadIdx.x;
    if (i < n) p[i] = 0.0f;
}

__global__ void k_embed(const int* __restrict__ ids,
                        const float* __restrict__ table,
                        float* __restrict__ out, int ntok, int D)
{
    int i = blockIdx.x * blockDim.x + threadIdx.x;
    if (i >= ntok * D) return;
    int t = i / D, d = i % D;
    out[i] = table[(size_t)ids[t] * D + d];
}

__device__ __forceinline__ float sigmoidf_(float x)
{
    return 1.0f / (1.0f + expf(-x));
}

// LSTM cell: gates (B x 4*Hd, PyTorch order i,f,g,o) -> update h,c in place.
// Optionally writes y (strided) and final h/c snapshots.
__global__ void k_cell(const float* __restrict__ g, long ldg,
                       float* __restrict__ h, float* __restrict__ c, int Hd,
                       float* __restrict__ y, long ldy,
                       float* __restrict__ hfin, float* __restrict__ cfin,
                       long ldfin, int writeFinal)
{
    int idx = blockIdx.x * blockDim.x + threadIdx.x;
    if (idx >= BB * Hd) return;
    int b = idx / Hd, j = idx - b * Hd;
    const float* gr = g + (size_t)b * (size_t)ldg;
    float gi = sigmoidf_(gr[j]);
    float gf = sigmoidf_(gr[Hd + j]);
    float gg = tanhf(gr[2 * Hd + j]);
    float go = sigmoidf_(gr[3 * Hd + j]);
    float cn = gf * c[idx] + gi * gg;
    float hn = go * tanhf(cn);
    c[idx] = cn;
    h[idx] = hn;
    if (y) y[(size_t)b * (size_t)ldy + j] = hn;
    if (writeFinal) {
        hfin[(size_t)b * (size_t)ldfin + j] = hn;
        cfin[(size_t)b * (size_t)ldfin + j] = cn;
    }
}

// dst[r][0:Da] = a[r]; dst[r][Da:Da+Db] = b[r]
__global__ void k_concat(float* __restrict__ dst, int ldd,
                         const float* __restrict__ a, long lda, int Da,
                         const float* __restrict__ b, long ldb, int Db, int rows)
{
    int D = Da + Db;
    int i = blockIdx.x * blockDim.x + threadIdx.x;
    if (i >= rows * D) return;
    int r = i / D, cidx = i - r * D;
    float v = (cidx < Da) ? a[(size_t)r * (size_t)lda + cidx]
                          : b[(size_t)r * (size_t)ldb + (cidx - Da)];
    dst[(size_t)r * ldd + cidx] = v;
}

// v = tanh(src); write to d1 (ld1) and d2 (ld2)
__global__ void k_tanh2(const float* __restrict__ src, int lds,
                        float* __restrict__ d1, int ld1,
                        float* __restrict__ d2, long ld2, int rows, int cols)
{
    int i = blockIdx.x * blockDim.x + threadIdx.x;
    if (i >= rows * cols) return;
    int r = i / cols, cidx = i - r * cols;
    float v = tanhf(src[(size_t)r * lds + cidx]);
    d1[(size_t)r * ld1 + cidx] = v;
    d2[(size_t)r * (size_t)ld2 + cidx] = v;
}

// Attention: one block per batch row. scores = q . h_src[s], softmax, ctx.
__global__ __launch_bounds__(256) void k_attn(const float* __restrict__ q,
                                              const float* __restrict__ hsrc,
                                              float* __restrict__ ctx)
{
    __shared__ float red[256];
    __shared__ float w[SS];
    __shared__ float lsh;
    const int b = blockIdx.x, tid = threadIdx.x;
    const float* qb = q + (size_t)b * HH;
    const float* hb = hsrc + (size_t)b * SS * HH;
    const int s = tid >> 2, p = tid & 3;   // 64 scores x 4 partial dots
    float acc = 0.0f;
    const int seg = HH / 4;
    for (int hh = p * seg; hh < (p + 1) * seg; ++hh)
        acc += qb[hh] * hb[(size_t)s * HH + hh];
    red[tid] = acc;
    __syncthreads();
    if (p == 0) w[s] = red[tid] + red[tid + 1] + red[tid + 2] + red[tid + 3];
    __syncthreads();
    if (tid == 0) {
        float m = -1e30f;
        for (int i = 0; i < SS; ++i) m = fmaxf(m, w[i]);
        float l = 0.0f;
        for (int i = 0; i < SS; ++i) { w[i] = expf(w[i] - m); l += w[i]; }
        lsh = l;
    }
    __syncthreads();
    const float invl = 1.0f / lsh;
    for (int hh = tid; hh < HH; hh += 256) {
        float a2 = 0.0f;
        for (int i = 0; i < SS; ++i) a2 += w[i] * hb[(size_t)i * HH + hh];
        ctx[(size_t)b * HH + hh] = a2 * invl;
    }
}

// In-place log-softmax along rows of length `cols`; one block per row.
__global__ __launch_bounds__(256) void k_logsoftmax(float* __restrict__ x, int cols)
{
    __shared__ float red[256];
    const int row = blockIdx.x, tid = threadIdx.x;
    float* xr = x + (size_t)row * cols;
    float m = -1e30f;
    for (int c = tid; c < cols; c += 256) m = fmaxf(m, xr[c]);
    red[tid] = m; __syncthreads();
    for (int off = 128; off > 0; off >>= 1) {
        if (tid < off) red[tid] = fmaxf(red[tid], red[tid + off]);
        __syncthreads();
    }
    m = red[0]; __syncthreads();
    float l = 0.0f;
    for (int c = tid; c < cols; c += 256) l += expf(xr[c] - m);
    red[tid] = l; __syncthreads();
    for (int off = 128; off > 0; off >>= 1) {
        if (tid < off) red[tid] += red[tid + off];
        __syncthreads();
    }
    const float lse = m + logf(red[0]);
    __syncthreads();
    for (int c = tid; c < cols; c += 256) xr[c] = xr[c] - lse;
}

// ---------------------------------------------------------------------------
// Host orchestration
// ---------------------------------------------------------------------------
extern "C" void kernel_launch(void* const* d_in, const int* in_sizes, int n_in,
                              void* d_out, int out_size, void* d_ws, size_t ws_size,
                              hipStream_t stream)
{
    (void)in_sizes; (void)n_in; (void)out_size; (void)ws_size;

    const int* src = (const int*)d_in[0];
    const int* tgt = (const int*)d_in[1];
    const float* embW_src = (const float*)d_in[2];
    const float* embW_dec = (const float*)d_in[3];
    // enc_params: base 4 + 6*l : [Wih_f, Whh_f, b_f, Wih_b, Whh_b, b_b]
    // dec_params: base 16 + 3*l: [Wih, Whh, b]
    const float* attn_W   = (const float*)d_in[22];
    const float* concat_W = (const float*)d_in[23];
    const float* concat_b = (const float*)d_in[24];
    const float* gen_W    = (const float*)d_in[25];
    const float* gen_b    = (const float*)d_in[26];
    float* out = (float*)d_out;

    // ---- workspace carve (floats) ----
    float* ws = (float*)d_ws;
    size_t off = 0;
    auto alloc = [&](size_t n) { float* p = ws + off; off += n; return p; };
    float* x0    = alloc((size_t)BB * SS * WVD);   // src embeddings
    float* x1    = alloc((size_t)BB * SS * HH);    // encoder layer0 output
    float* x2    = alloc((size_t)BB * SS * HH);    // encoder layer1 output = h_src
    float* gbf   = alloc((size_t)BB * SS * 4 * HD2); // fwd gates (x@Wih+b, += h@Whh)
    float* gbb   = alloc((size_t)BB * SS * 4 * HD2); // bwd gates
    float* embt  = alloc((size_t)BB * TT * WVD);   // tgt embeddings
    float* htseq = alloc((size_t)BB * TT * HH);    // decoder h_tilde sequence
    float* h_enc = alloc((size_t)BB * HD2);
    float* c_enc = alloc((size_t)BB * HD2);
    float* dh0   = alloc((size_t)BB * HH);
    float* dc0   = alloc((size_t)BB * HH);
    float* dh1   = alloc((size_t)BB * HH);
    float* dc1   = alloc((size_t)BB * HH);
    float* htil  = alloc((size_t)BB * HH);
    float* xcat  = alloc((size_t)BB * (WVD + HH));
    float* catc  = alloc((size_t)BB * 2 * HH);
    float* gates = alloc((size_t)BB * 4 * HH);
    float* qbuf  = alloc((size_t)BB * HH);
    float* ctxb  = alloc((size_t)BB * HH);

    auto gemm = [&](const float* A, int lda, const float* W, int ldw,
                    const float* bias, float* C, long ldc,
                    int M, int N, int K, int accum) {
        int tiles = (M / 16) * (N / 64);   // 16x64 strips per wave
        int blocks = (tiles + 7) / 8;
        k_gemm<<<blocks, 256, 0, stream>>>(A, lda, W, ldw, bias, C, ldc,
                                           N, K, accum, tiles);
    };
    auto zero = [&](float* p, int n) {
        k_zero<<<(n + 255) / 256, 256, 0, stream>>>(p, n);
    };

    // ---- embeddings ----
    {
        int n = BB * SS * WVD;
        k_embed<<<(n + 255) / 256, 256, 0, stream>>>(src, embW_src, x0, BB * SS, WVD);
        n = BB * TT * WVD;
        k_embed<<<(n + 255) / 256, 256, 0, stream>>>(tgt, embW_dec, embt, BB * TT, WVD);
    }

    // ---- encoder: 2 layers, bidirectional ----
    const float* xin = x0;
    int din = WVD;
    float* xout = x1;
    for (int l = 0; l < 2; ++l) {
        const float* Wihf = (const float*)d_in[4 + 6 * l + 0];
        const float* Whhf = (const float*)d_in[4 + 6 * l + 1];
        const float* bf   = (const float*)d_in[4 + 6 * l + 2];
        const float* Wihb = (const float*)d_in[4 + 6 * l + 3];
        const float* Whhb = (const float*)d_in[4 + 6 * l + 4];
        const float* bb   = (const float*)d_in[4 + 6 * l + 5];
        float* hfin = (l == 0) ? dh0 : dh1;
        float* cfin = (l == 0) ? dc0 : dc1;

        // time-parallel input transforms: (B*S, 4*HD) = X @ Wih^T + b
        gemm(xin, din, Wihf, din, bf, gbf, 4 * HD2, BB * SS, 4 * HD2, din, 0);
        gemm(xin, din, Wihb, din, bb, gbb, 4 * HD2, BB * SS, 4 * HD2, din, 0);

        const long ldg = (long)SS * 4 * HD2;   // batch-row stride at fixed t
        const long ldy = (long)SS * HH;

        // forward direction
        zero(h_enc, BB * HD2); zero(c_enc, BB * HD2);
        for (int t = 0; t < SS; ++t) {
            gemm(h_enc, HD2, Whhf, HD2, nullptr,
                 gbf + (size_t)t * 4 * HD2, ldg, BB, 4 * HD2, HD2, 1);
            int n = BB * HD2;
            k_cell<<<(n + 255) / 256, 256, 0, stream>>>(
                gbf + (size_t)t * 4 * HD2, ldg, h_enc, c_enc, HD2,
                xout + (size_t)t * HH + 0, ldy,
                hfin + 0, cfin + 0, (long)HH, (t == SS - 1) ? 1 : 0);
        }
        // backward direction
        zero(h_enc, BB * HD2); zero(c_enc, BB * HD2);
        for (int t = SS - 1; t >= 0; --t) {
            gemm(h_enc, HD2, Whhb, HD2, nullptr,
                 gbb + (size_t)t * 4 * HD2, ldg, BB, 4 * HD2, HD2, 1);
            int n = BB * HD2;
            k_cell<<<(n + 255) / 256, 256, 0, stream>>>(
                gbb + (size_t)t * 4 * HD2, ldg, h_enc, c_enc, HD2,
                xout + (size_t)t * HH + HD2, ldy,
                hfin + HD2, cfin + HD2, (long)HH, (t == 0) ? 1 : 0);
        }
        xin = xout; din = HH; xout = x2;
    }
    const float* h_src = x2;

    // ---- decoder with attention + input feeding ----
    const float* Wih0 = (const float*)d_in[16];
    const float* Whh0 = (const float*)d_in[17];
    const float* b0   = (const float*)d_in[18];
    const float* Wih1 = (const float*)d_in[19];
    const float* Whh1 = (const float*)d_in[20];
    const float* b1   = (const float*)d_in[21];

    zero(htil, BB * HH);
    const int ncell = BB * HH;
    for (int t = 0; t < TT; ++t) {
        // xt = [emb_tgt[:,t,:], h_tilde]
        {
            int n = BB * (WVD + HH);
            k_concat<<<(n + 255) / 256, 256, 0, stream>>>(
                xcat, WVD + HH, embt + (size_t)t * WVD, (long)TT * WVD, WVD,
                htil, (long)HH, HH, BB);
        }
        // layer 0
        gemm(xcat, WVD + HH, Wih0, WVD + HH, b0, gates, 4 * HH, BB, 4 * HH, WVD + HH, 0);
        gemm(dh0, HH, Whh0, HH, nullptr, gates, 4 * HH, BB, 4 * HH, HH, 1);
        k_cell<<<(ncell + 255) / 256, 256, 0, stream>>>(
            gates, (long)(4 * HH), dh0, dc0, HH,
            nullptr, 0, nullptr, nullptr, 0, 0);
        // layer 1
        gemm(dh0, HH, Wih1, HH, b1, gates, 4 * HH, BB, 4 * HH, HH, 0);
        gemm(dh1, HH, Whh1, HH, nullptr, gates, 4 * HH, BB, 4 * HH, HH, 1);
        k_cell<<<(ncell + 255) / 256, 256, 0, stream>>>(
            gates, (long)(4 * HH), dh1, dc1, HH,
            nullptr, 0, nullptr, nullptr, 0, 0);
        // attention
        gemm(dh1, HH, attn_W, HH, nullptr, qbuf, HH, BB, HH, HH, 0);
        k_attn<<<BB, 256, 0, stream>>>(qbuf, h_src, ctxb);
        // h_tilde = tanh([dec_out, ctx] @ concat_W^T + b)
        {
            int n = BB * 2 * HH;
            k_concat<<<(n + 255) / 256, 256, 0, stream>>>(
                catc, 2 * HH, dh1, (long)HH, HH, ctxb, (long)HH, HH, BB);
        }
        gemm(catc, 2 * HH, concat_W, 2 * HH, concat_b, gates, HH, BB, HH, 2 * HH, 0);
        {
            int n = BB * HH;
            k_tanh2<<<(n + 255) / 256, 256, 0, stream>>>(
                gates, HH, htil, HH, htseq + (size_t)t * HH, (long)TT * HH, BB, HH);
        }
    }

    // ---- generator: logits + log-softmax ----
    gemm(htseq, HH, gen_W, HH, gen_b, out, VV, BB * TT, VV, HH, 0);
    k_logsoftmax<<<BB * TT, 256, 0, stream>>>(out, VV);
}
